// HierarchicalMemory_84799834293121
// MI455X (gfx1250) — compile-verified, matrix operand order checked by reading, not run on hardware
//
#include <hip/hip_runtime.h>
#include <hip/hip_bf16.h>
#include <math.h>

// ---------------------------------------------------------------------------
// Problem constants (from reference): B=8, D_IN=1024, DIM=256, H=8, HD=32,
// SIZES={32768,8192,2048}, NUM_LEVELS=3
// ---------------------------------------------------------------------------
#define NB    8
#define DIN   1024
#define DIM   256
#define NH    8
#define HD    32
#define NLVL  3
#define CHUNK 64            // mem rows per block in the WMMA kernel

typedef __bf16 bf16;
typedef __attribute__((ext_vector_type(16))) __bf16        v16bf;
typedef __attribute__((ext_vector_type(8)))  float         v8f;
typedef __attribute__((ext_vector_type(4)))  unsigned int  v4u;
typedef __attribute__((ext_vector_type(8)))  int           v8i_;
typedef __attribute__((ext_vector_type(4)))  int           v4i_;

// ---------------------------------------------------------------------------
// 1) p[lvl][b][c] = x[b] . proj_w[lvl][c] + proj_b[lvl][c]
// ---------------------------------------------------------------------------
__global__ __launch_bounds__(256)
void hm_proj_kernel(const float* __restrict__ x, const float* __restrict__ pw,
                    const float* __restrict__ pb, float* __restrict__ p_ws) {
    int idx = blockIdx.x * 256 + threadIdx.x;          // 3*8*256 = 6144
    if (idx >= NLVL * NB * DIM) return;
    int lvl = idx / (NB * DIM);
    int r   = idx % (NB * DIM);
    int b   = r / DIM;
    int c   = r % DIM;
    const float* xr = x + (size_t)b * DIN;
    const float* wr = pw + (size_t)lvl * DIM * DIN + (size_t)c * DIN;
    float acc = pb[lvl * DIM + c];
    for (int d = 0; d < DIN; ++d) acc += xr[d] * wr[d];
    p_ws[idx] = acc;
}

// ---------------------------------------------------------------------------
// 2) q[lvl][b][c] = p[lvl][b] . wq[lvl][c] + bq[lvl][c]   (wq = inw[lvl][0:256])
// ---------------------------------------------------------------------------
__global__ __launch_bounds__(256)
void hm_qproj_kernel(const float* __restrict__ p_ws, const float* __restrict__ inw,
                     const float* __restrict__ inb, float* __restrict__ q_ws) {
    int idx = blockIdx.x * 256 + threadIdx.x;
    if (idx >= NLVL * NB * DIM) return;
    int lvl = idx / (NB * DIM);
    int r   = idx % (NB * DIM);
    int b   = r / DIM;
    int c   = r % DIM;
    const float* pr = p_ws + (size_t)(lvl * NB + b) * DIM;
    const float* wr = inw + (size_t)lvl * 3 * DIM * DIM + (size_t)c * DIM;
    float acc = inb[lvl * 3 * DIM + c];
    for (int d = 0; d < DIM; ++d) acc += pr[d] * wr[d];
    q_ws[idx] = acc;
}

// ---------------------------------------------------------------------------
// 3) Fused K/V projection (bf16 WMMA) + partial flash attention per 64-row
//    chunk of the memory bank.  The fp32 mem chunk (64 rows x 256 cols,
//    64 KB contiguous) is staged into LDS by the Tensor Data Mover.
//    W      = inw[lvl] rows 256..767  ([wk;wv], 512 x 256, row-major fp32)
//    biasKV = inb[lvl] + 256          (512 floats)
//    q      = q_ws[lvl]               (8 x 256)
//    part   : per chunk, per (b,h) pair: {m, l, o[32]} = 34 floats
// ---------------------------------------------------------------------------
__global__ __launch_bounds__(256)
void hm_kv_attn_kernel(const float* __restrict__ mem,
                       const float* __restrict__ W,
                       const float* __restrict__ biasKV,
                       const float* __restrict__ q,
                       float* __restrict__ part,
                       int Sdim) {
    extern __shared__ char smem_raw[];
    float (*s_memf)[DIM]   = (float(*)[DIM])smem_raw;                         // 64 KB
    float (*s_kv)[2 * DIM] = (float(*)[2 * DIM])(smem_raw + CHUNK * DIM * 4); // 128 KB

    const int tid   = threadIdx.x;
    const int chunk = blockIdx.x;
    const int s0    = chunk * CHUNK;

    // ---- TDM: async 2-D tile load (64 rows x 256 f32) into LDS ----------
    if (tid < 32) {
        unsigned long long ga =
            (unsigned long long)(uintptr_t)(mem + (size_t)s0 * DIM);
        v4u g0;
        g0[0] = 1u;                                          // count=1, user desc
        g0[1] = 0u;                                          // lds_addr = 0 (dyn LDS base)
        g0[2] = (unsigned)(ga & 0xffffffffull);              // global_addr[31:0]
        g0[3] = (unsigned)((ga >> 32) & 0x1ffffffull) | 0x80000000u; // addr[56:32] | type=2
        v8i_ g1;
        g1[0] = 0x20000;                                     // data_size = 2 (4 bytes)
        g1[1] = (int)(256u << 16);                           // tensor_dim0[15:0] = 256
        g1[2] = (int)(((unsigned)Sdim & 0xffffu) << 16);     // tensor_dim1[15:0]
        g1[3] = (int)((((unsigned)Sdim >> 16) & 0xffffu) | (256u << 16)); // dim1 hi | tile_dim0=256
        g1[4] = CHUNK;                                       // tile_dim1 = 64, tile_dim2 = 0
        g1[5] = 256;                                         // tensor_dim0_stride[31:0] = 256
        g1[6] = 0;
        g1[7] = 0;
        v4i_ g2 = {0, 0, 0, 0};
        v4i_ g3 = {0, 0, 0, 0};
        v8i_ g4 = {0, 0, 0, 0, 0, 0, 0, 0};
        __builtin_amdgcn_tensor_load_to_lds(g0, g1, g2, g3, g4, 0);
        __builtin_amdgcn_s_wait_tensorcnt(0);
    }
    __syncthreads();

    // ---- WMMA: [64 x 512] = chunk(bf16) @ W^T(bf16), fp32 accum ---------
    const int wave = tid >> 5;
    const int lane = tid & 31;
    const int half = lane >> 4;      // 0: lanes 0-15, 1: lanes 16-31
    const int lr   = lane & 15;

    // wave -> M-tile (wave&3) and 16 of the 32 N-tiles ((wave>>2)*16 ..)
    const int m0  = (wave & 3) * 16;
    const int nt0 = (wave >> 2) * 16;

    // A fragments held in registers, reused across all 16 N-tiles.
    v16bf afrag[8];
    {
        const float* arow = &s_memf[m0 + lr][0];
#pragma unroll
        for (int ks = 0; ks < 8; ++ks) {
            const int ka = ks * 32 + half * 8;
#pragma unroll
            for (int j = 0; j < 8; ++j) {
                afrag[ks][j]     = (bf16)arow[ka + j];        // K = ka..ka+7
                afrag[ks][8 + j] = (bf16)arow[ka + 16 + j];   // K = ka+16..ka+23
            }
        }
    }

    for (int nt = nt0; nt < nt0 + 16; ++nt) {
        const int n0 = nt * 16;
        const float* wrow = W + (size_t)(n0 + lr) * DIM;
        __builtin_prefetch(wrow + 16 * DIM, 0, 1);           // next N-tile's row
        v8f acc = {0.f, 0.f, 0.f, 0.f, 0.f, 0.f, 0.f, 0.f};
#pragma unroll
        for (int ks = 0; ks < 8; ++ks) {
            const int ka = ks * 32 + half * 8;
            v16bf bfrag;
#pragma unroll
            for (int j = 0; j < 8; ++j) {
                bfrag[j]     = (bf16)wrow[ka + j];
                bfrag[8 + j] = (bf16)wrow[ka + 16 + j];
            }
            acc = __builtin_amdgcn_wmma_f32_16x16x32_bf16(
                false, afrag[ks], false, bfrag, (short)0, acc, false, false);
        }
        // D layout: lane lr -> N = n0+lr ; VGPR r -> M = m0 + r + 8*half
        const float bn = biasKV[n0 + lr];
#pragma unroll
        for (int r = 0; r < 8; ++r)
            s_kv[m0 + half * 8 + r][n0 + lr] = acc[r] + bn;
    }
    __syncthreads();

    // ---- partial single-query attention over this chunk -----------------
    // 64 (b,h) pairs, 4 lanes per pair, 16 keys per lane.
    const float scale = 0.1767766952966369f;   // 1/sqrt(32)
    const int pair = tid >> 2;                 // 0..63
    const int b_   = pair >> 3;
    const int h    = pair & 7;
    const int sub  = tid & 3;
    const int sb   = sub * 16;

    float qreg[HD];
    const float* qp = q + (size_t)b_ * DIM + h * HD;
    for (int d = 0; d < HD; ++d) qreg[d] = qp[d];

    float scr[16];
    float m = -1e30f;
    for (int j = 0; j < 16; ++j) {
        const float* kr = &s_kv[sb + j][h * HD];
        float sc = 0.f;
        for (int d = 0; d < HD; ++d) sc += qreg[d] * kr[d];
        sc *= scale;
        scr[j] = sc;
        m = fmaxf(m, sc);
    }
    m = fmaxf(m, __shfl_xor(m, 1, 32));
    m = fmaxf(m, __shfl_xor(m, 2, 32));

    float l = 0.f;
    float oacc[HD];
    for (int d = 0; d < HD; ++d) oacc[d] = 0.f;
    for (int j = 0; j < 16; ++j) {
        float e = __expf(scr[j] - m);
        l += e;
        const float* vr = &s_kv[sb + j][DIM + h * HD];
        for (int d = 0; d < HD; ++d) oacc[d] += e * vr[d];
    }
    l += __shfl_xor(l, 1, 32);
    l += __shfl_xor(l, 2, 32);
    for (int d = 0; d < HD; ++d) {
        oacc[d] += __shfl_xor(oacc[d], 1, 32);
        oacc[d] += __shfl_xor(oacc[d], 2, 32);
    }
    if (sub == 0) {
        float* P = part + ((size_t)chunk * 64 + pair) * 34;
        P[0] = m;
        P[1] = l;
        for (int d = 0; d < HD; ++d) P[2 + d] = oacc[d];
    }
}

// ---------------------------------------------------------------------------
// 4) log-sum-exp combine of chunk partials -> o[lvl][b][h*32+d]
// ---------------------------------------------------------------------------
__global__ __launch_bounds__(32)
void hm_attn_combine_kernel(const float* __restrict__ part, float* __restrict__ o_ws) {
    const int lvl  = blockIdx.x / 64;
    const int pair = blockIdx.x % 64;
    const int d    = threadIdx.x;
    const int nch  = (lvl == 0) ? 512 : (lvl == 1) ? 128 : 32;
    const size_t off = (lvl == 0) ? 0 : (lvl == 1) ? (size_t)512 * 2176 : (size_t)640 * 2176;
    const float* P0 = part + off + (size_t)pair * 34;

    float m = -1e30f, l = 0.f, acc = 0.f;
    for (int c = 0; c < nch; ++c) {
        const float* P = P0 + (size_t)c * 2176;   // 64 pairs * 34 floats per chunk
        float mc = P[0], lc = P[1], oc = P[2 + d];
        float mn = fmaxf(m, mc);
        float f1 = __expf(m - mn), f2 = __expf(mc - mn);
        l   = l * f1 + lc * f2;
        acc = acc * f1 + oc * f2;
        m   = mn;
    }
    int b = pair >> 3, h = pair & 7;
    o_ws[(size_t)(lvl * NB + b) * DIM + h * HD + d] = acc / l;
}

// ---------------------------------------------------------------------------
// 5) op[lvl][b][c] = o[lvl][b] . outw[lvl][c] + outb[lvl][c]
// ---------------------------------------------------------------------------
__global__ __launch_bounds__(256)
void hm_outproj_kernel(const float* __restrict__ o_ws, const float* __restrict__ ow,
                       const float* __restrict__ ob, float* __restrict__ op_ws) {
    int idx = blockIdx.x * 256 + threadIdx.x;
    if (idx >= NLVL * NB * DIM) return;
    int lvl = idx / (NB * DIM);
    int r   = idx % (NB * DIM);
    int b   = r / DIM;
    int c   = r % DIM;
    const float* orow = o_ws + (size_t)(lvl * NB + b) * DIM;
    const float* wrow = ow + (size_t)lvl * DIM * DIM + (size_t)c * DIM;
    float acc = ob[lvl * DIM + c];
    for (int d = 0; d < DIM; ++d) acc += orow[d] * wrow[d];
    op_ws[idx] = acc;
}

// ---------------------------------------------------------------------------
// 6) out[b][j] = concat(op)[b] . merg_w[j] + merg_b[j]
// ---------------------------------------------------------------------------
__global__ __launch_bounds__(256)
void hm_merge_kernel(const float* __restrict__ op_ws, const float* __restrict__ mw,
                     const float* __restrict__ mb, float* __restrict__ out) {
    int idx = blockIdx.x * 256 + threadIdx.x;          // 8*1024 = 8192
    if (idx >= NB * DIN) return;
    int b = idx / DIN;
    int j = idx % DIN;
    const float* wrow = mw + (size_t)j * (NLVL * DIM);
    float acc = mb[j];
    for (int t = 0; t < NLVL * DIM; ++t) {
        int lvl = t >> 8, c = t & (DIM - 1);
        acc += wrow[t] * op_ws[(size_t)(lvl * NB + b) * DIM + c];
    }
    out[idx] = acc;
}

// ---------------------------------------------------------------------------
// 7) updated_mem[b][s][:] = mem[s][:] + p[lvl][b][:]   (float4 streaming)
// ---------------------------------------------------------------------------
__global__ __launch_bounds__(256)
void hm_mem_update_kernel(const float* __restrict__ mem, const float* __restrict__ p,
                          float* __restrict__ out, int S) {
    __shared__ float4 s_p[NB * (DIM / 4)];             // 512 float4 = 8 KB
    const float4* p4 = (const float4*)p;
    for (int i = threadIdx.x; i < NB * (DIM / 4); i += 256) s_p[i] = p4[i];
    __syncthreads();

    size_t idx = (size_t)blockIdx.x * 256 + threadIdx.x;
    size_t n4  = (size_t)NB * S * (DIM / 4);
    if (idx >= n4) return;
    size_t d4 = idx & 63;                              // DIM/4 = 64 (pow2)
    size_t s  = (idx >> 6) % (size_t)S;                // S is pow2
    size_t b  = idx / ((size_t)S * 64);
    float4 mv = ((const float4*)mem)[s * 64 + d4];
    float4 pv = s_p[b * 64 + d4];
    float4 ov;
    ov.x = mv.x + pv.x; ov.y = mv.y + pv.y; ov.z = mv.z + pv.z; ov.w = mv.w + pv.w;
    ((float4*)out)[idx] = ov;
}

// ---------------------------------------------------------------------------
extern "C" void kernel_launch(void* const* d_in, const int* in_sizes, int n_in,
                              void* d_out, int out_size, void* d_ws, size_t ws_size,
                              hipStream_t stream) {
    const float* x      = (const float*)d_in[0];
    const float* proj_w = (const float*)d_in[1];
    const float* proj_b = (const float*)d_in[2];
    const float* mem0   = (const float*)d_in[3];
    const float* mem1   = (const float*)d_in[4];
    const float* mem2   = (const float*)d_in[5];
    const float* inw    = (const float*)d_in[6];
    const float* inb    = (const float*)d_in[7];
    const float* outw   = (const float*)d_in[8];
    const float* outb   = (const float*)d_in[9];
    const float* merg_w = (const float*)d_in[10];
    const float* merg_b = (const float*)d_in[11];
    float* out = (float*)d_out;
    float* ws  = (float*)d_ws;

    // workspace layout (floats)
    float* p_ws  = ws;              // 6144
    float* q_ws  = ws + 6144;       // 6144
    float* o_ws  = ws + 12288;      // 6144
    float* op_ws = ws + 18432;      // 6144
    float* part  = ws + 24576;      // (512+128+32)*64*34 = 1,462,272

    const float* mems[NLVL] = {mem0, mem1, mem2};
    const int    S[NLVL]    = {32768, 8192, 2048};
    const size_t part_off[NLVL] = {0, (size_t)512 * 2176, (size_t)640 * 2176};
    const size_t out_off[NLVL]  = {
        (size_t)NB * DIN,
        (size_t)NB * DIN + (size_t)NB * 32768 * DIM,
        (size_t)NB * DIN + (size_t)NB * 32768 * DIM + (size_t)NB * 8192 * DIM};

    hm_proj_kernel <<<24, 256, 0, stream>>>(x, proj_w, proj_b, p_ws);
    hm_qproj_kernel<<<24, 256, 0, stream>>>(p_ws, inw, inb, q_ws);

    const size_t lds_bytes = (size_t)CHUNK * DIM * sizeof(float)
                           + (size_t)CHUNK * 2 * DIM * sizeof(float); // 192 KB
    for (int lvl = 0; lvl < NLVL; ++lvl) {
        hm_kv_attn_kernel<<<S[lvl] / CHUNK, 256, lds_bytes, stream>>>(
            mems[lvl],
            inw + (size_t)lvl * 3 * DIM * DIM + (size_t)DIM * DIM,  // [wk;wv]
            inb + lvl * 3 * DIM + DIM,                              // [bk;bv]
            q_ws + (size_t)lvl * NB * DIM,
            part + part_off[lvl],
            S[lvl]);
    }

    hm_attn_combine_kernel<<<NLVL * 64, 32, 0, stream>>>(part, o_ws);
    hm_outproj_kernel<<<24, 256, 0, stream>>>(o_ws, outw, outb, op_ws);
    hm_merge_kernel  <<<32, 256, 0, stream>>>(op_ws, merg_w, merg_b, out);

    for (int lvl = 0; lvl < NLVL; ++lvl) {
        size_t n4 = (size_t)NB * S[lvl] * (DIM / 4);
        hm_mem_update_kernel<<<(unsigned)((n4 + 255) / 256), 256, 0, stream>>>(
            mems[lvl], p_ws + (size_t)lvl * NB * DIM, out + out_off[lvl], S[lvl]);
    }
}
